// C_Cross_Attention3D_6820408066841
// MI455X (gfx1250) — matrix operand order repaired and checked by reading, not compiled, not run on hardware
//
#include <hip/hip_runtime.h>
#include <hip/hip_bf16.h>

// ---- types for WMMA ----
typedef __attribute__((ext_vector_type(16))) __bf16 v16bf;
typedef __attribute__((ext_vector_type(8)))  float  v8f;

#define NTPOS 128          // positions per workgroup in main GEMM kernel
#define XLSTRIDE 784       // bf16 elems per LDS row (768 + 16 pad, keeps 32B alignment)

__device__ __forceinline__ unsigned short f32_to_bf16(float f) {
    unsigned int u = __float_as_uint(f);
    unsigned int r = u + 0x7FFFu + ((u >> 16) & 1u);   // round-to-nearest-even
    return (unsigned short)(r >> 16);
}
__device__ __forceinline__ float bf16_to_f32(unsigned short u) {
    return __uint_as_float(((unsigned int)u) << 16);
}

// ---------------- K0: zero the atomic accumulators ----------------
__global__ __launch_bounds__(256) void k0_zero(float* __restrict__ p, long long n) {
    long long i = (long long)blockIdx.x * 256 + threadIdx.x;
    if (i < n) p[i] = 0.0f;
}

// ---------------- K1: pre-swizzle kv_w (f32) -> bf16 WMMA A-fragments ----------------
// A 16x32 bf16 fragment lane layout (contiguous, matches B): lanes 0-15 row M=lane, elem j -> K=j;
// lanes 16-31 row M=lane-16, elem j -> K=16+j.
__global__ __launch_bounds__(32) void k1_wswz(const float* __restrict__ kv_w,
                                              unsigned short* __restrict__ Wswz) {
    int blk = blockIdx.x;             // strip*24 + kb, 96*24 blocks
    int strip = blk / 24, kb = blk % 24;
    int lane = threadIdx.x;
    int row  = lane & 15;
    int hiK  = (lane >> 4) * 16;
    unsigned short* dst = Wswz + (long long)blk * 512 + lane * 16;
    const float* src = kv_w + (long long)(strip * 16 + row) * 768 + kb * 32;
    #pragma unroll
    for (int j = 0; j < 16; ++j) {
        int k = hiK + j;
        dst[j] = f32_to_bf16(src[k]);
    }
}

// ---------------- K2: q = q_w@y + q_b, then per-head L2 normalize ----------------
__global__ __launch_bounds__(256) void k2_q(const float* __restrict__ y,
                                            const float* __restrict__ q_w,
                                            const float* __restrict__ q_b,
                                            float* __restrict__ qhatG) {
    int b = blockIdx.x, t = threadIdx.x;
    __shared__ float ys[768], qs[768], nrm[12];
    for (int i = t; i < 768; i += 256) ys[i] = y[b * 768 + i];
    __syncthreads();
    for (int o = t; o < 768; o += 256) {
        float s = q_b[o];
        const float* wr = q_w + (long long)o * 768;
        for (int c = 0; c < 768; ++c) s += wr[c] * ys[c];
        qs[o] = s;
    }
    __syncthreads();
    if (t < 12) {
        float s = 0.f;
        for (int d = 0; d < 64; ++d) { float v = qs[t * 64 + d]; s += v * v; }
        nrm[t] = fmaxf(sqrtf(s), 1e-12f);
    }
    __syncthreads();
    for (int o = t; o < 768; o += 256) qhatG[b * 768 + o] = qs[o] / nrm[o >> 6];
}

// ---------------- K3: main fused GEMM (bf16 WMMA) + k-stats scatter + v store ----------------
__global__ __launch_bounds__(256) void k3_gemm(const float* __restrict__ x,
                                               const float* __restrict__ kv_b,
                                               const unsigned short* __restrict__ Wswz,
                                               const float* __restrict__ qhatG,
                                               float* __restrict__ ssqG,
                                               float* __restrict__ qdotG,
                                               unsigned short* __restrict__ Vbuf) {
    extern __shared__ char smem[];
    unsigned short* xl = (unsigned short*)smem;                       // 128 x 784 bf16
    float* qh    = (float*)(smem + NTPOS * XLSTRIDE * 2);             // 768
    float* biass = qh + 768;                                          // 1536
    float* lssq  = biass + 1536;                                      // 1536*2
    float* lqdot = lssq + 3072;                                       // 1536*2

    int wg   = blockIdx.x;                 // 0..511
    int b    = wg >> 8;                    // batch
    int n0   = (wg & 255) * NTPOS;         // conv-position window base
    int tid  = threadIdx.x;
    int lane = tid & 31;
    int wave = tid >> 5;                   // 8 waves; wave == posblk

    for (int i = tid; i < 768;  i += 256) qh[i] = qhatG[b * 768 + i];
    for (int i = tid; i < 1536; i += 256) biass[i] = kv_b[i];
    for (int i = tid; i < 3072; i += 256) { lssq[i] = 0.f; lqdot[i] = 0.f; }

    // stage x window -> bf16 LDS, transposed: xl[pos][channel]
    {
        int c_sub = tid >> 5;              // 0..7
        int pq    = tid & 31;              // quad of positions
        const float* xb = x + (long long)b * 768 * 32768 + n0;
        for (int cg = 0; cg < 96; ++cg) {
            int c = cg * 8 + c_sub;
            float4 v = *(const float4*)(xb + (long long)c * 32768 + pq * 4);
            int p = pq * 4;
            xl[(p + 0) * XLSTRIDE + c] = f32_to_bf16(v.x);
            xl[(p + 1) * XLSTRIDE + c] = f32_to_bf16(v.y);
            xl[(p + 2) * XLSTRIDE + c] = f32_to_bf16(v.z);
            xl[(p + 3) * XLSTRIDE + c] = f32_to_bf16(v.w);
        }
    }
    __syncthreads();

    int posblk = wave;                       // 16-position sub-tile
    int col    = lane & 15;                  // C/D fragment column = position
    int lh     = lane >> 4;                  // lane half -> row half
    int koff   = (lane < 16) ? 0 : 16;       // B fragment: lanes 0-15 K=0..15, 16-31 K=16..31
    const unsigned short* Bbase = xl + (posblk * 16 + col) * XLSTRIDE + koff;

    for (int strip = 0; strip < 96; ++strip) {
        // two independent accumulator chains to break the XDL RAW dependency
        v8f acc0 = {};
        v8f acc1 = {};
        const unsigned short* Aptr = Wswz + (long long)strip * 24 * 512 + lane * 16;
        #pragma unroll 4
        for (int kb = 0; kb < 24; kb += 2) {
            v16bf a0 = *(const v16bf*)(Aptr + kb * 512);
            v16bf b0 = *(const v16bf*)(Bbase + kb * 32);
            v16bf a1 = *(const v16bf*)(Aptr + (kb + 1) * 512);
            v16bf b1 = *(const v16bf*)(Bbase + (kb + 1) * 32);
            acc0 = __builtin_amdgcn_wmma_f32_16x16x32_bf16(
                false, a0, false, b0, (short)0, acc0, false, false);
            acc1 = __builtin_amdgcn_wmma_f32_16x16x32_bf16(
                false, a1, false, b1, (short)0, acc1, false, false);
        }
        // epilogue: rows of this 16x16 tile; each 16-wide row segment is wholly k or v,
        // with uniform (n', head).
        #pragma unroll
        for (int vg = 0; vg < 8; ++vg) {
            int M  = vg + 8 * lh;
            int c2 = strip * 16 + M;
            long long j0 = (long long)c2 * 32768 + n0 + posblk * 16;
            int r0  = (int)(j0 % 1536);
            int npr = (int)(j0 / 1536);
            float val = acc0[vg] + acc1[vg] + biass[c2];
            if (r0 < 768) {                       // k segment
                float ps = val * val;
                float pq = qh[r0 + col] * val;
                #pragma unroll
                for (int m = 1; m < 16; m <<= 1) {
                    ps += __shfl_xor(ps, m, 32);
                    pq += __shfl_xor(pq, m, 32);
                }
                if (col == 0) {
                    int sub = (r0 >> 6) & 1;      // head-half within 128 window
                    atomicAdd(&lssq[c2 * 2 + sub], ps);
                    atomicAdd(&lqdot[c2 * 2 + sub], pq);
                }
            } else {                               // v segment -> scrambled bf16 store
                long long off = ((long long)(b * 32768 + npr)) * 768 + (r0 - 768) + col;
                Vbuf[off] = f32_to_bf16(val);
            }
        }
    }
    __syncthreads();

    // flush per-(channel, head-half) k stats to global per-(b,h,n') arrays
    for (int c2 = tid; c2 < 1536; c2 += 256) {
        long long j0 = (long long)c2 * 32768 + n0;
        int r0 = (int)(j0 % 1536);
        if (r0 < 768) {
            int npr = (int)(j0 / 1536);
            int h0  = r0 >> 6;
            #pragma unroll
            for (int sub = 0; sub < 2; ++sub) {
                long long gi = ((long long)(b * 12 + h0 + sub)) * 32768 + npr;
                atomicAdd(&ssqG[gi],  lssq[c2 * 2 + sub]);
                atomicAdd(&qdotG[gi], lqdot[c2 * 2 + sub]);
            }
        }
    }
}

// ---------------- K4: softmax weights (logits bounded in [-1,1] -> no max pass) ----------------
__global__ __launch_bounds__(256) void k4_w(const float* __restrict__ ssqG,
                                            const float* __restrict__ qdotG,
                                            float* __restrict__ wG,
                                            float* __restrict__ gden) {
    long long i = (long long)blockIdx.x * 256 + threadIdx.x;   // < 2*12*32768
    float nrm = fmaxf(sqrtf(ssqG[i]), 1e-12f);
    float w = __expf(qdotG[i] / nrm);
    wG[i] = w;
    __shared__ float red[256];
    red[threadIdx.x] = w;
    __syncthreads();
    for (int s = 128; s > 0; s >>= 1) {
        if (threadIdx.x < s) red[threadIdx.x] += red[threadIdx.x + s];
        __syncthreads();
    }
    if (threadIdx.x == 0) atomicAdd(&gden[i >> 15], red[0]);   // i>>15 == b*12+h
}

// ---------------- K5: out_attn numerator = sum_n' w[b,h,n'] * v[b,n',c] ----------------
__global__ __launch_bounds__(256) void k5_acc(const unsigned short* __restrict__ Vbuf,
                                              const float* __restrict__ wG,
                                              float* __restrict__ gacc) {
    int blk = blockIdx.x;               // 128 blocks: 64 per batch x 512 n' each
    int b   = blk >> 6;
    int p0  = (blk & 63) * 512;
    int t   = threadIdx.x;
    int c0 = t, c1 = t + 256, c2 = t + 512;
    float a0 = 0.f, a1 = 0.f, a2 = 0.f;
    const float* w0 = wG + ((long long)b * 12 + (c0 >> 6)) * 32768;
    const float* w1 = wG + ((long long)b * 12 + (c1 >> 6)) * 32768;
    const float* w2 = wG + ((long long)b * 12 + (c2 >> 6)) * 32768;
    const unsigned short* vb = Vbuf + (long long)b * 32768 * 768;
    for (int p = p0; p < p0 + 512; ++p) {
        const unsigned short* vr = vb + (long long)p * 768;
        a0 += w0[p] * bf16_to_f32(vr[c0]);
        a1 += w1[p] * bf16_to_f32(vr[c1]);
        a2 += w2[p] * bf16_to_f32(vr[c2]);
    }
    atomicAdd(&gacc[b * 768 + c0], a0);
    atomicAdd(&gacc[b * 768 + c1], a1);
    atomicAdd(&gacc[b * 768 + c2], a2);
}

// ---------------- K6: divide by denom, final projection + bias ----------------
__global__ __launch_bounds__(256) void k6_proj(const float* __restrict__ gacc,
                                               const float* __restrict__ gden,
                                               const float* __restrict__ proj_w,
                                               const float* __restrict__ proj_b,
                                               float* __restrict__ out) {
    int b = blockIdx.x, t = threadIdx.x;
    __shared__ float att[768];
    for (int i = t; i < 768; i += 256)
        att[i] = gacc[b * 768 + i] / gden[b * 12 + (i >> 6)];
    __syncthreads();
    for (int o = t; o < 768; o += 256) {
        float s = proj_b[o];
        const float* wr = proj_w + (long long)o * 768;
        for (int c = 0; c < 768; ++c) s += wr[c] * att[c];
        out[b * 768 + o] = s;
    }
}

extern "C" void kernel_launch(void* const* d_in, const int* in_sizes, int n_in,
                              void* d_out, int out_size, void* d_ws, size_t ws_size,
                              hipStream_t stream) {
    const float* x      = (const float*)d_in[0];
    const float* y      = (const float*)d_in[1];
    const float* q_w    = (const float*)d_in[2];
    const float* q_b    = (const float*)d_in[3];
    const float* kv_w   = (const float*)d_in[4];
    const float* kv_b   = (const float*)d_in[5];
    const float* proj_w = (const float*)d_in[6];
    const float* proj_b = (const float*)d_in[7];
    float* out = (float*)d_out;

    // workspace layout (floats unless noted)
    float* ssqG  = (float*)d_ws;                 // 2*12*32768
    float* qdotG = ssqG  + 786432;               // 2*12*32768
    float* gacc  = qdotG + 786432;               // 2*768
    float* gden  = gacc  + 1536;                 // 2*12
    float* qhatG = gden  + 24;                   // 2*768
    float* wG    = qhatG + 1536;                 // 2*12*32768
    unsigned short* Wswz = (unsigned short*)(wG + 786432);   // 96*24*512 bf16 (32B aligned)
    unsigned short* Vbuf = Wswz + 1179648;                   // 2*32768*768 bf16

    long long nzero = 786432LL * 2 + 1536 + 24;  // ssqG|qdotG|gacc|gden contiguous
    k0_zero<<<(int)((nzero + 255) / 256), 256, 0, stream>>>(ssqG, nzero);
    k1_wswz<<<96 * 24, 32, 0, stream>>>(kv_w, Wswz);
    k2_q<<<2, 256, 0, stream>>>(y, q_w, q_b, qhatG);

    size_t smem3 = (size_t)NTPOS * XLSTRIDE * 2        // x tile (bf16)
                 + (768 + 1536 + 3072 + 3072) * 4;     // qhat, bias, lssq, lqdot
    k3_gemm<<<512, 256, smem3, stream>>>(x, kv_b, Wswz, qhatG, ssqG, qdotG, Vbuf);

    k4_w<<<3072, 256, 0, stream>>>(ssqG, qdotG, wG, gden);
    k5_acc<<<128, 256, 0, stream>>>(Vbuf, wG, gacc);
    k6_proj<<<2, 256, 0, stream>>>(gacc, gden, proj_w, proj_b, out);
}
